// DependencyEncoder_7456063226107
// MI455X (gfx1250) — compile-verified, matrix operand order checked by reading, not run on hardware
//
#include <hip/hip_runtime.h>

// ---------------- Types for CDNA5 WMMA ----------------
typedef __attribute__((ext_vector_type(16))) __bf16 v16bf;
typedef __attribute__((ext_vector_type(8)))  __bf16 v8bf;
typedef __attribute__((ext_vector_type(8)))  float  v8f;
typedef __bf16 bf16_t;

// ---------------- Problem dims ----------------
#define BB   128   // batch
#define NTOK 128   // tokens N
#define TT   256   // transitions T
#define HH   256   // hidden H
#define DD   50    // tracking D
#define DP   64    // padded D
#define G4   200   // 4*D
#define GP   208   // padded 4*D (13 tiles of 16)
#define K3H  768   // 3*H

#define ROWS_PER_WG 16
#define NUM_WG      (BB / ROWS_PER_WG)   // 8
#define BLOCK       256                  // 8 waves

// ---------------- WMMA helpers ----------------
__device__ __forceinline__ v8f wmma_bf16(v16bf a, v16bf b, v8f c) {
  // 8 args: (neg_a, A, neg_b, B, c_mod, C, reuse_a, reuse_b)
  return __builtin_amdgcn_wmma_f32_16x16x32_bf16(false, a, false, b, (short)0, c,
                                                 false, false);
}

// Load a 16x32 bf16 fragment (A layout; B uses same layout with N in place of M).
// Memory layout: base[row][k] with k contiguous, row stride = ld elements.
// Per ISA: lanes 0-15 hold row=lane, K={k0..k0+7, k0+16..k0+23};
//          lanes 16-31 hold row=lane-16, K={k0+8..k0+15, k0+24..k0+31}.
__device__ __forceinline__ v16bf load_frag_bf(const bf16_t* base, int ld, int k0, int lane) {
  int l = lane & 15;
  const bf16_t* p = base + (size_t)l * ld + k0 + ((lane & 16) ? 8 : 0);
  v8bf lo = *(const v8bf*)(p);
  v8bf hi = *(const v8bf*)(p + 16);
  return __builtin_shufflevector(lo, hi, 0, 1, 2, 3, 4, 5, 6, 7,
                                 8, 9, 10, 11, 12, 13, 14, 15);
}

__device__ __forceinline__ float sigf(float x) { return 1.f / (1.f + __expf(-x)); }

// ---------------- Prep: fp32 -> bf16 weights ([N][K], K-padded), zero stack slot 0 ----------------
__global__ void prep_kernel(const float* __restrict__ Wx,   // (4,256,50)
                            const float* __restrict__ Ul,   // (4,256,256)
                            const float* __restrict__ Ur,   // (4,256,256)
                            const float* __restrict__ Wih,  // (200,768)
                            const float* __restrict__ Whh,  // (200,50)
                            bf16_t* __restrict__ Ul_bf,     // (4,256,256)
                            bf16_t* __restrict__ Ur_bf,     // (4,256,256)
                            bf16_t* __restrict__ Wih_bf,    // (208,768)
                            bf16_t* __restrict__ Whh_bf,    // (208,64)
                            bf16_t* __restrict__ Wx_bf,     // (4,256,64)
                            float* __restrict__ stack_h,    // (128,257,256)
                            float* __restrict__ stack_c) {
  int tid = blockIdx.x * blockDim.x + threadIdx.x;
  int stride = gridDim.x * blockDim.x;
  for (int i = tid; i < 4 * 256 * 256; i += stride) {
    Ul_bf[i] = (bf16_t)Ul[i];
    Ur_bf[i] = (bf16_t)Ur[i];
  }
  for (int i = tid; i < GP * K3H; i += stride) {
    int n = i / K3H, k = i % K3H;
    Wih_bf[i] = (n < G4) ? (bf16_t)Wih[n * K3H + k] : (bf16_t)0.f;
  }
  for (int i = tid; i < GP * DP; i += stride) {
    int n = i / DP, k = i % DP;
    Whh_bf[i] = (n < G4 && k < DD) ? (bf16_t)Whh[n * DD + k] : (bf16_t)0.f;
  }
  for (int i = tid; i < 4 * 256 * DP; i += stride) {
    int g = i / (256 * DP), r = (i / DP) % 256, k = i % DP;
    Wx_bf[i] = (k < DD) ? (bf16_t)Wx[(g * 256 + r) * DD + k] : (bf16_t)0.f;
  }
  // zero stack slot 0 for every batch row (read when queue is empty)
  for (int i = tid; i < BB * HH; i += stride) {
    int b = i / HH, c = i % HH;
    size_t idx = ((size_t)b * (TT + 1)) * HH + c;
    stack_h[idx] = 0.f;
    stack_c[idx] = 0.f;
  }
}

// ---------------- Main persistent scan kernel ----------------
__global__ __launch_bounds__(BLOCK, 1) void spinn_scan_kernel(
    const float* __restrict__ tokens_h,   // (B,N,H)
    const float* __restrict__ tokens_c,   // (B,N,H)
    const int*   __restrict__ transitions,// (B,T)
    const float* __restrict__ h0,         // (B,D)
    const float* __restrict__ c0,         // (B,D)
    const float* __restrict__ bgate,      // (4,H)
    const float* __restrict__ bih,        // (200)
    const float* __restrict__ bhh,        // (200)
    const bf16_t* __restrict__ Ul_bf,     // (4,256,256) [g][n][k]
    const bf16_t* __restrict__ Ur_bf,
    const bf16_t* __restrict__ Wih_bf,    // (208,768)
    const bf16_t* __restrict__ Whh_bf,    // (208,64)
    const bf16_t* __restrict__ Wx_bf,     // (4,256,64)
    float* __restrict__ stack_h,          // (B,T+1,H)
    float* __restrict__ stack_c,
    int*   __restrict__ queue_g,          // (B,T+1)
    float* __restrict__ out)              // (B,H)
{
  __shared__ bf16_t bt_bf[ROWS_PER_WG][HH];
  __shared__ bf16_t h1_bf[ROWS_PER_WG][HH];
  __shared__ bf16_t h2_bf[ROWS_PER_WG][HH];
  __shared__ bf16_t c1_bf[ROWS_PER_WG][HH];
  __shared__ bf16_t c2_bf[ROWS_PER_WG][HH];
  __shared__ bf16_t th_bf[ROWS_PER_WG][DP];
  __shared__ float  tc_f [ROWS_PER_WG][DP];
  __shared__ float  gates[ROWS_PER_WG][G4];
  __shared__ int sp1[ROWS_PER_WG], sp2[ROWS_PER_WG], bp[ROWS_PER_WG];
  __shared__ int trs[ROWS_PER_WG], qlen[ROWS_PER_WG], bptr[ROWS_PER_WG];

  const int tid  = threadIdx.x;
  const int lane = tid & 31;
  const int wave = tid >> 5;
  const int b0   = blockIdx.x * ROWS_PER_WG;

  // ---- init per-call state (deterministic across replays) ----
  if (tid < ROWS_PER_WG) { qlen[tid] = 0; bptr[tid] = 0; }
  for (int i = tid; i < ROWS_PER_WG * DP; i += BLOCK) {
    int r = i / DP, c = i % DP;
    float hv = (c < DD) ? h0[(b0 + r) * DD + c] : 0.f;
    float cv = (c < DD) ? c0[(b0 + r) * DD + c] : 0.f;
    th_bf[r][c] = (bf16_t)hv;
    tc_f [r][c] = cv;
  }
  __syncthreads();

  for (int t = 1; t <= TT; ++t) {
    // ---- phase 0: per-row stack/queue pointer logic ----
    if (tid < ROWS_PER_WG) {
      int r = tid;
      int q = qlen[r];
      int trans = transitions[(b0 + r) * TT + (t - 1)];
      trs[r] = trans;
      const int* Q = queue_g + (size_t)(b0 + r) * (TT + 1);
      sp1[r] = (q >= 1) ? Q[q - 1] : 0;
      sp2[r] = (q >= 2) ? Q[q - 2] : 0;
      int bpv = bptr[r];
      bp[r] = (bpv < NTOK - 1) ? bpv : (NTOK - 1);
    }
    __syncthreads();

    // ---- phase 1: gather state rows into LDS (fp32 -> bf16) ----
    for (int i = tid; i < ROWS_PER_WG * HH; i += BLOCK) {
      int r = i >> 8, c = i & 255;
      int brow = b0 + r;
      size_t base = (size_t)brow * (TT + 1) * HH;
      bt_bf[r][c] = (bf16_t)tokens_h[((size_t)brow * NTOK + bp[r]) * HH + c];
      h1_bf[r][c] = (bf16_t)stack_h[base + (size_t)sp1[r] * HH + c];
      h2_bf[r][c] = (bf16_t)stack_h[base + (size_t)sp2[r] * HH + c];
      c1_bf[r][c] = (bf16_t)stack_c[base + (size_t)sp1[r] * HH + c];
      c2_bf[r][c] = (bf16_t)stack_c[base + (size_t)sp2[r] * HH + c];
    }
    __syncthreads();

    // ---- phase 2: tracking-LSTM GEMM: [bt_h|h1|h2] @ Wih^T + th @ Whh^T ----
    for (int tile = wave; tile < GP / 16; tile += 8) {
      int n0 = tile * 16;
      v8f acc = {};
      const bf16_t* Wn = Wih_bf + (size_t)n0 * K3H;
      for (int kc = 0; kc < HH; kc += 32) {
        v16bf a0 = load_frag_bf(&bt_bf[0][0], HH, kc, lane);
        acc = wmma_bf16(a0, load_frag_bf(Wn, K3H, kc, lane), acc);
        v16bf a1 = load_frag_bf(&h1_bf[0][0], HH, kc, lane);
        acc = wmma_bf16(a1, load_frag_bf(Wn, K3H, kc + 256, lane), acc);
        v16bf a2 = load_frag_bf(&h2_bf[0][0], HH, kc, lane);
        acc = wmma_bf16(a2, load_frag_bf(Wn, K3H, kc + 512, lane), acc);
      }
      const bf16_t* Whn = Whh_bf + (size_t)n0 * DP;
      for (int kc = 0; kc < DP; kc += 32) {
        v16bf a = load_frag_bf(&th_bf[0][0], DP, kc, lane);
        acc = wmma_bf16(a, load_frag_bf(Whn, DP, kc, lane), acc);
      }
      int nn = lane & 15, mb = (lane & 16) ? 8 : 0;
      int col = n0 + nn;
      if (col < G4) {
        float bb = bih[col] + bhh[col];
#pragma unroll
        for (int r = 0; r < 8; ++r) gates[mb + r][col] = acc[r] + bb;
      }
    }
    __syncthreads();

    // ---- phase 3: elementwise tracking LSTM (torch gate order i,f,g,o) ----
    for (int i = tid; i < ROWS_PER_WG * DP; i += BLOCK) {
      int r = i / DP, c = i % DP;
      if (c < DD) {
        float gi = gates[r][c];
        float gf = gates[r][c + DD];
        float gg = gates[r][c + 2 * DD];
        float go = gates[r][c + 3 * DD];
        float tc = sigf(gf) * tc_f[r][c] + sigf(gi) * tanhf(gg);
        float th = sigf(go) * tanhf(tc);
        tc_f [r][c] = tc;
        th_bf[r][c] = (bf16_t)th;
      } else {
        th_bf[r][c] = (bf16_t)0.f;
      }
    }
    __syncthreads();

    // ---- phase 4: two tree-cell calls, 32 column-tile tasks ----
    // call0: tree_cell(th, h1, h2, c1, c2) -> Ul in = c1, Ur in = h1, c_l = c2, c_r = h2
    // call1: tree_cell(th, h2, h1, c2, c2) -> Ul in = c2, Ur in = h2, c_l = c2, c_r = h1
    for (int task = wave; task < 32; task += 8) {
      int call = task >> 4;
      int n0 = (task & 15) * 16;
      const bf16_t* Aul = call ? &c2_bf[0][0] : &c1_bf[0][0];
      const bf16_t* Aur = call ? &h2_bf[0][0] : &h1_bf[0][0];

      v8f ai = {}, ao = {}, au = {};
      // x projections: i uses Wx[0], o and f use Wx[2], u uses Wx[3]
      for (int kc = 0; kc < DP; kc += 32) {
        v16bf a = load_frag_bf(&th_bf[0][0], DP, kc, lane);
        ai = wmma_bf16(a, load_frag_bf(Wx_bf + ((size_t)0 * 256 + n0) * DP, DP, kc, lane), ai);
        ao = wmma_bf16(a, load_frag_bf(Wx_bf + ((size_t)2 * 256 + n0) * DP, DP, kc, lane), ao);
        au = wmma_bf16(a, load_frag_bf(Wx_bf + ((size_t)3 * 256 + n0) * DP, DP, kc, lane), au);
      }
      v8f af = ao;  // f gate shares the xo projection (faithful to source)

      for (int kc = 0; kc < HH; kc += 32) {
        v16bf a = load_frag_bf(Aul, HH, kc, lane);
        ai = wmma_bf16(a, load_frag_bf(Ul_bf + ((size_t)0 * 256 + n0) * HH, HH, kc, lane), ai);
        af = wmma_bf16(a, load_frag_bf(Ul_bf + ((size_t)1 * 256 + n0) * HH, HH, kc, lane), af);
        ao = wmma_bf16(a, load_frag_bf(Ul_bf + ((size_t)2 * 256 + n0) * HH, HH, kc, lane), ao);
        au = wmma_bf16(a, load_frag_bf(Ul_bf + ((size_t)3 * 256 + n0) * HH, HH, kc, lane), au);
      }
      for (int kc = 0; kc < HH; kc += 32) {
        v16bf a = load_frag_bf(Aur, HH, kc, lane);
        ai = wmma_bf16(a, load_frag_bf(Ur_bf + ((size_t)0 * 256 + n0) * HH, HH, kc, lane), ai);
        af = wmma_bf16(a, load_frag_bf(Ur_bf + ((size_t)1 * 256 + n0) * HH, HH, kc, lane), af);
        ao = wmma_bf16(a, load_frag_bf(Ur_bf + ((size_t)2 * 256 + n0) * HH, HH, kc, lane), ao);
        au = wmma_bf16(a, load_frag_bf(Ur_bf + ((size_t)3 * 256 + n0) * HH, HH, kc, lane), au);
      }

      // elementwise gate math on the C/D layout; predicated store by transition type
      int nn = lane & 15, mb = (lane & 16) ? 8 : 0;
      int col = n0 + nn;
      float b_i = bgate[0 * HH + col];
      float b_f = bgate[1 * HH + col];
      float b_o = bgate[2 * HH + col];
      float b_u = bgate[3 * HH + col];
#pragma unroll
      for (int r = 0; r < 8; ++r) {
        int m = mb + r;
        int trans = trs[m];
        bool write = (call == 0) ? (trans == 1) : (trans >= 2);
        if (!write) continue;
        int brow = b0 + m;
        size_t base = (size_t)brow * (TT + 1) * HH;
        float gi = sigf(ai[r] + b_i);
        float gf = sigf(af[r] + b_f);
        float go = sigf(ao[r] + b_o);
        float gu = tanhf(au[r] + b_u);
        float cl = stack_c[base + (size_t)sp2[m] * HH + col];                       // c2
        float cr = call ? stack_h[base + (size_t)sp1[m] * HH + col]                // h1
                        : stack_h[base + (size_t)sp2[m] * HH + col];               // h2
        float cj = gi * gu + gf * cl + gf * cr;
        float hj = go * tanhf(cj);
        stack_h[base + (size_t)t * HH + col] = hj;
        stack_c[base + (size_t)t * HH + col] = cj;
        if (t == TT) out[(size_t)brow * HH + col] = hj;
      }
    }

    // ---- shift rows (trans==0): push token onto stack ----
    for (int i = tid; i < ROWS_PER_WG * HH; i += BLOCK) {
      int r = i >> 8, c = i & 255;
      if (trs[r] == 0) {
        int brow = b0 + r;
        size_t tbase = ((size_t)brow * NTOK + bp[r]) * HH + c;
        float hv = tokens_h[tbase];
        float cv = tokens_c[tbase];
        size_t sbase = (size_t)brow * (TT + 1) * HH + (size_t)t * HH + c;
        stack_h[sbase] = hv;
        stack_c[sbase] = cv;
        if (t == TT) out[(size_t)brow * HH + c] = hv;
      }
    }
    __threadfence();
    __syncthreads();

    // ---- phase 5: update queue / qlen / bptr ----
    if (tid < ROWS_PER_WG) {
      int r = tid;
      int mask = (trs[r] >= 1);
      int q = qlen[r];
      int q2 = mask ? ((q - 2 > 0) ? (q - 2) : 0) : q;
      queue_g[(size_t)(b0 + r) * (TT + 1) + q2] = t;
      qlen[r] = q2 + 1;
      bptr[r] += mask ? 0 : 1;
    }
    __syncthreads();
  }
}

// ---------------- Launch ----------------
extern "C" void kernel_launch(void* const* d_in, const int* in_sizes, int n_in,
                              void* d_out, int out_size, void* d_ws, size_t ws_size,
                              hipStream_t stream) {
  (void)in_sizes; (void)n_in; (void)out_size; (void)ws_size;
  const float* tokens_h    = (const float*)d_in[0];
  const float* tokens_c    = (const float*)d_in[1];
  const int*   transitions = (const int*)d_in[2];
  const float* h0          = (const float*)d_in[3];
  const float* c0          = (const float*)d_in[4];
  const float* Wx          = (const float*)d_in[5];
  const float* Ul          = (const float*)d_in[6];
  const float* Ur          = (const float*)d_in[7];
  const float* bgate       = (const float*)d_in[8];
  const float* Wih         = (const float*)d_in[9];
  const float* Whh         = (const float*)d_in[10];
  const float* bih         = (const float*)d_in[11];
  const float* bhh         = (const float*)d_in[12];
  float* out = (float*)d_out;

  char* ws = (char*)d_ws;
  size_t off = 0;
  auto take = [&](size_t bytes) -> char* {
    char* p = ws + off;
    off = (off + bytes + 255) & ~(size_t)255;
    return p;
  };
  bf16_t* Ul_bf  = (bf16_t*)take((size_t)4 * 256 * 256 * sizeof(bf16_t));
  bf16_t* Ur_bf  = (bf16_t*)take((size_t)4 * 256 * 256 * sizeof(bf16_t));
  bf16_t* Wih_bf = (bf16_t*)take((size_t)GP * K3H * sizeof(bf16_t));
  bf16_t* Whh_bf = (bf16_t*)take((size_t)GP * DP * sizeof(bf16_t));
  bf16_t* Wx_bf  = (bf16_t*)take((size_t)4 * 256 * DP * sizeof(bf16_t));
  int*    queue  = (int*)take((size_t)BB * (TT + 1) * sizeof(int));
  float*  stk_h  = (float*)take((size_t)BB * (TT + 1) * HH * sizeof(float));
  float*  stk_c  = (float*)take((size_t)BB * (TT + 1) * HH * sizeof(float));

  prep_kernel<<<512, 256, 0, stream>>>(Wx, Ul, Ur, Wih, Whh,
                                       Ul_bf, Ur_bf, Wih_bf, Whh_bf, Wx_bf,
                                       stk_h, stk_c);
  spinn_scan_kernel<<<NUM_WG, BLOCK, 0, stream>>>(
      tokens_h, tokens_c, transitions, h0, c0, bgate, bih, bhh,
      Ul_bf, Ur_bf, Wih_bf, Whh_bf, Wx_bf,
      stk_h, stk_c, queue, out);
}
